// DeepGCN_63548336112227
// MI455X (gfx1250) — compile-verified, matrix operand order checked by reading, not run on hardware
//
#include <hip/hip_runtime.h>
#include <hip/hip_bf16.h>

// ---------------------------------------------------------------------------
// DeepGCN (Vision-GNN + prompts) forward for MI455X / gfx1250.
// All 1x1 convs, patch-MLP, kNN-similarity and head run through a bf16 WMMA
// GEMM (v_wmma_f32_16x16x32_bf16, f32 accumulate, fused bias/BN-affine/act).
// 3x3 convs are direct kernels (small FLOP share). Workspace is a hand-planned
// arena (~193 MB) with phase-wise slot reuse.
// Input pointer order assumed: depth-first insertion order of setup_inputs().
// ---------------------------------------------------------------------------

typedef __attribute__((ext_vector_type(16))) __bf16 v16bf;
typedef __attribute__((ext_vector_type(8)))  float  v8f;

__device__ __forceinline__ unsigned short f2bf(float f) {
  union { float f; unsigned int u; } x; x.f = f;
  unsigned int r = x.u + 0x7FFFu + ((x.u >> 16) & 1u);   // round-to-nearest-even
  return (unsigned short)(r >> 16);
}

union Frag { v16bf v; unsigned short u[16]; };

#define GT_M 64
#define GT_N 64
#define GT_K 32

// C[M,N] = act( affine( A[M,K] * B[K,N] + bias ) ), all f32 in memory,
// bf16 WMMA compute. biasMode: 0 none, 1 per-row(M), 2 per-col(N).
// act: 0 none, 1 relu, 2 leaky(0.01). g/bt per-row affine if g != nullptr.
__global__ __launch_bounds__(128)
void k_gemm(const float* __restrict__ A, const float* __restrict__ Bm,
            float* __restrict__ C, int M, int N, int K,
            long long sA, long long sB, long long sC,
            const float* __restrict__ bias, int biasMode,
            const float* __restrict__ g, const float* __restrict__ bt, int act)
{
  __shared__ unsigned short As[GT_M][GT_K + 2];
  __shared__ unsigned short Bs[GT_K][GT_N + 2];
  const int tid  = threadIdx.x;
  const int wave = tid >> 5;
  const int lane = tid & 31;
  A  += (long long)blockIdx.z * sA;
  Bm += (long long)blockIdx.z * sB;
  C  += (long long)blockIdx.z * sC;
  const int nBase = blockIdx.x * GT_N;
  const int mBase = blockIdx.y * GT_M;
  const int m0w = (wave >> 1) * 32;
  const int n0w = (wave & 1) * 32;
  v8f acc[2][2] = {};

  for (int k0 = 0; k0 < K; k0 += GT_K) {
    // Stage A tile (64x32) and B tile (32x64) as bf16 in LDS (16 elems/thread)
    #pragma unroll
    for (int i = 0; i < 16; ++i) {
      int e = tid * 16 + i;
      int r = e >> 5, c = e & 31;
      int gm = mBase + r, gk = k0 + c;
      float v = (gm < M && gk < K) ? A[(long long)gm * K + gk] : 0.0f;
      As[r][c] = f2bf(v);
    }
    #pragma unroll
    for (int i = 0; i < 16; ++i) {
      int e = tid * 16 + i;
      int r = e >> 6, c = e & 63;
      int gk = k0 + r, gn = nBase + c;
      float v = (gk < K && gn < N) ? Bm[(long long)gk * N + gn] : 0.0f;
      Bs[r][c] = f2bf(v);
    }
    __syncthreads();

    Frag a[2], b[2];
    const int half = lane >> 4;
    #pragma unroll
    for (int mi = 0; mi < 2; ++mi) {
      int row = m0w + mi * 16 + (lane & 15);
      #pragma unroll
      for (int r = 0; r < 8; ++r) {
        int kk = ((r & 4) << 2) + half * 8 + (r & 3) * 2;
        a[mi].u[2 * r]     = As[row][kk];
        a[mi].u[2 * r + 1] = As[row][kk + 1];
      }
    }
    #pragma unroll
    for (int ni = 0; ni < 2; ++ni) {
      int col = n0w + ni * 16 + (lane & 15);
      #pragma unroll
      for (int r = 0; r < 8; ++r) {
        int kk = ((r & 4) << 2) + half * 8 + (r & 3) * 2;
        b[ni].u[2 * r]     = Bs[kk][col];
        b[ni].u[2 * r + 1] = Bs[kk + 1][col];
      }
    }
    #pragma unroll
    for (int mi = 0; mi < 2; ++mi)
      #pragma unroll
      for (int ni = 0; ni < 2; ++ni)
        acc[mi][ni] = __builtin_amdgcn_wmma_f32_16x16x32_bf16(
            false, a[mi].v, false, b[ni].v, (short)0, acc[mi][ni], false, false);
    __syncthreads();
  }

  // Epilogue: C/D layout per ISA: VGPR r -> row r + 8*(lane/16), col lane%16
  #pragma unroll
  for (int mi = 0; mi < 2; ++mi)
    #pragma unroll
    for (int ni = 0; ni < 2; ++ni)
      #pragma unroll
      for (int r = 0; r < 8; ++r) {
        int row = mBase + m0w + mi * 16 + r + 8 * (lane >> 4);
        int col = nBase + n0w + ni * 16 + (lane & 15);
        if (row < M && col < N) {
          float v = acc[mi][ni][r];
          if (biasMode == 1)      v += bias[row];
          else if (biasMode == 2) v += bias[col];
          if (g) v = v * g[row] + bt[row];
          if (act == 1)      v = fmaxf(v, 0.0f);
          else if (act == 2) v = v > 0.0f ? v : 0.01f * v;
          C[(long long)row * N + col] = v;
        }
      }
}

// ------------------------- direct 3x3 conv (fused bias/affine/act) ---------
__global__ void k_conv3x3(const float* __restrict__ in, const float* __restrict__ w,
                          const float* __restrict__ bias, const float* __restrict__ g,
                          const float* __restrict__ bt, float* __restrict__ out,
                          int Bn, int Ci, int H, int W, int Co, int Ho, int Wo,
                          int stride, int act)
{
  long long i = blockIdx.x * 256LL + threadIdx.x;
  long long total = (long long)Bn * Co * Ho * Wo;
  if (i >= total) return;
  int x = (int)(i % Wo); long long t = i / Wo;
  int y = (int)(t % Ho); t /= Ho;
  int co = (int)(t % Co); int b = (int)(t / Co);
  float s = bias ? bias[co] : 0.0f;
  const float* wp = w + (long long)co * Ci * 9;
  const float* ip = in + (long long)b * Ci * H * W;
  int y0 = y * stride - 1, x0 = x * stride - 1;
  for (int ci = 0; ci < Ci; ++ci) {
    const float* ic = ip + (long long)ci * H * W;
    const float* wc = wp + ci * 9;
    #pragma unroll
    for (int ky = 0; ky < 3; ++ky) {
      int yy = y0 + ky; if (yy < 0 || yy >= H) continue;
      #pragma unroll
      for (int kx = 0; kx < 3; ++kx) {
        int xx = x0 + kx; if (xx < 0 || xx >= W) continue;
        s += wc[ky * 3 + kx] * ic[(long long)yy * W + xx];
      }
    }
  }
  if (g) s = s * g[co] + bt[co];
  if (act == 1)      s = fmaxf(s, 0.0f);
  else if (act == 2) s = s > 0.0f ? s : 0.01f * s;
  out[i] = s;
}

// ------------------------- pooling -----------------------------------------
__global__ void k_maxpool(const float* __restrict__ in, float* __restrict__ out,
                          int BC, int H, int W, int offY, int offX, int k,
                          int Ho, int Wo)
{
  long long i = blockIdx.x * 256LL + threadIdx.x;
  long long total = (long long)BC * Ho * Wo;
  if (i >= total) return;
  int x = (int)(i % Wo); long long t = i / Wo;
  int y = (int)(t % Ho); int bc = (int)(t / Ho);
  const float* ip = in + (long long)bc * H * W;
  float m = -3.4e38f;
  for (int dy = 0; dy < k; ++dy)
    for (int dx = 0; dx < k; ++dx)
      m = fmaxf(m, ip[(long long)(offY + y * k + dy) * W + offX + x * k + dx]);
  out[i] = m;
}

__global__ void k_avgpool(const float* __restrict__ in, float* __restrict__ out,
                          int BC, int H, int W, int k, int Ho, int Wo)
{
  long long i = blockIdx.x * 256LL + threadIdx.x;
  long long total = (long long)BC * Ho * Wo;
  if (i >= total) return;
  int x = (int)(i % Wo); long long t = i / Wo;
  int y = (int)(t % Ho); int bc = (int)(t / Ho);
  const float* ip = in + (long long)bc * H * W;
  float s = 0.0f;
  for (int dy = 0; dy < k; ++dy)
    for (int dx = 0; dx < k; ++dx)
      s += ip[(long long)(y * k + dy) * W + x * k + dx];
  out[i] = s / (float)(k * k);
}

// ------------------------- prompt patch gather / combine --------------------
__global__ void k_patches(const float* __restrict__ img, float* __restrict__ A)
{
  long long i = blockIdx.x * 256LL + threadIdx.x;
  if (i >= 3136LL * 768LL) return;
  int col = (int)(i % 768); int row = (int)(i / 768);
  int b = row / 196, p = row % 196, py = p / 14, px = p % 14;
  int c = col / 256, q = col % 256, iy = q / 16, ix = q % 16;
  A[i] = img[(((long long)b * 3 + c) * 224 + py * 16 + iy) * 224 + px * 16 + ix];
}

__global__ void k_make_x0(const float* __restrict__ img, const float* __restrict__ P1,
                          const float* __restrict__ P2, float* __restrict__ x0)
{
  long long i = blockIdx.x * 256LL + threadIdx.x;
  long long total = 16LL * 3 * 224 * 224;
  if (i >= total) return;
  int x = (int)(i % 224); long long t = i / 224;
  int y = (int)(t % 224); t /= 224;
  int c = (int)(t % 3); int b = (int)(t / 3);
  int py = y >> 4, iy = y & 15, px = x >> 4, ix = x & 15;
  long long row = ((long long)b * 14 + py) * 14 + px;
  long long col = (long long)c * 256 + iy * 16 + ix;
  x0[i] = img[i] + 0.02f * (P1[row * 768 + col] + P2[i]);
}

// ------------------------- grapher helpers ----------------------------------
__global__ void k_concat(const float* __restrict__ x1, const float* __restrict__ np,
                         const float* __restrict__ tp, float* __restrict__ xp,
                         int C, int H, int W, int useInst)
{
  long long i = blockIdx.x * 256LL + threadIdx.x;
  int Hp = H + 4;
  long long total = 16LL * C * Hp * W;
  if (i >= total) return;
  int x = (int)(i % W); long long t = i / W;
  int h = (int)(t % Hp); t /= Hp;
  int c = (int)(t % C); int b = (int)(t / C);
  float v;
  if (h < H) {
    v = x1[(((long long)b * C + c) * H + h) * W + x];
  } else {
    v = np[((long long)c * 4 + (h - H)) * W + x];
    if (useInst) v += tp[((long long)b * 14 + x) * 256 + c];
  }
  xp[i] = v;
}

__global__ void k_l2t(const float* __restrict__ x, float* __restrict__ At,
                      int C, int N)
{
  long long i = blockIdx.x * 256LL + threadIdx.x;
  if (i >= 16LL * N) return;
  int n = (int)(i % N); int b = (int)(i / N);
  const float* col = x + (long long)b * C * N + n;
  float s = 0.0f;
  for (int c = 0; c < C; ++c) { float v = col[(long long)c * N]; s += v * v; }
  float inv = 1.0f / fmaxf(sqrtf(s), 1e-12f);
  float* o = At + ((long long)b * N + n) * C;
  for (int c = 0; c < C; ++c) o[c] = col[(long long)c * N] * inv;
}

__global__ void k_l2(const float* __restrict__ x, float* __restrict__ o,
                     int C, int Mm)
{
  long long i = blockIdx.x * 256LL + threadIdx.x;
  if (i >= 16LL * Mm) return;
  int m = (int)(i % Mm); int b = (int)(i / Mm);
  const float* col = x + (long long)b * C * Mm + m;
  float s = 0.0f;
  for (int c = 0; c < C; ++c) { float v = col[(long long)c * Mm]; s += v * v; }
  float inv = 1.0f / fmaxf(sqrtf(s), 1e-12f);
  float* oc = o + (long long)b * C * Mm + m;
  for (int c = 0; c < C; ++c) oc[(long long)c * Mm] = col[(long long)c * Mm] * inv;
}

__global__ void k_topk(const float* __restrict__ S, int* __restrict__ idx,
                       int N, int Mm)
{
  long long i = blockIdx.x * 256LL + threadIdx.x;
  if (i >= 16LL * N) return;
  const float* row = S + i * Mm;
  float bv[9]; int bi[9];
  #pragma unroll
  for (int j = 0; j < 9; ++j) { bv[j] = -3.4e38f; bi[j] = 0; }
  for (int m = 0; m < Mm; ++m) {
    float v = row[m];
    if (v > bv[8]) {
      int j = 8;
      while (j > 0 && bv[j - 1] < v) { bv[j] = bv[j - 1]; bi[j] = bi[j - 1]; --j; }
      bv[j] = v; bi[j] = m;
    }
  }
  int* o = idx + i * 9;
  #pragma unroll
  for (int j = 0; j < 9; ++j) o[j] = bi[j];
}

// max-relative gather, interleaved output (2c -> x, 2c+1 -> max_k(y_j - x))
__global__ void k_grmax(const float* __restrict__ xp, const float* __restrict__ y,
                        const int* __restrict__ idx, float* __restrict__ H2,
                        int C, int N, int Mm)
{
  long long i = blockIdx.x * 256LL + threadIdx.x;
  long long total = 16LL * C * N;
  if (i >= total) return;
  int n = (int)(i % N); long long t = i / N;
  int c = (int)(t % C); int b = (int)(t / C);
  float xv = xp[(((long long)b * C + c) * N) + n];
  const int* id = idx + ((long long)b * N + n) * 9;
  const float* yc = y + ((long long)b * C + c) * Mm;
  float mx = -3.4e38f;
  #pragma unroll
  for (int k = 0; k < 9; ++k) mx = fmaxf(mx, yc[id[k]] - xv);
  float* o = H2 + (long long)b * (2 * C) * N;
  o[(long long)(2 * c) * N + n]     = xv;
  o[(long long)(2 * c + 1) * N + n] = mx;
}

__global__ void k_cropres(const float* __restrict__ X2, const float* __restrict__ sc,
                          float* __restrict__ xg, int C, int H, int W)
{
  long long i = blockIdx.x * 256LL + threadIdx.x;
  long long total = 16LL * C * H * W;
  if (i >= total) return;
  int x = (int)(i % W); long long t = i / W;
  int h = (int)(t % H); t /= H;
  int c = (int)(t % C); int b = (int)(t / C);
  int Hp = H + 4;
  xg[i] = X2[(((long long)b * C + c) * Hp + h) * W + x] + sc[i];
}

// ------------------------- misc elementwise ---------------------------------
__global__ void k_add(const float* __restrict__ a, const float* __restrict__ b,
                      float* __restrict__ o, long long n)
{
  long long i = blockIdx.x * 256LL + threadIdx.x;
  if (i < n) o[i] = a[i] + b[i];
}

__global__ void k_addpe(float* __restrict__ x, const float* __restrict__ pe,
                        long long n, long long per)
{
  long long i = blockIdx.x * 256LL + threadIdx.x;
  if (i < n) x[i] += pe[i % per];
}

__global__ void k_ln(const float* __restrict__ x, const float* __restrict__ g,
                     const float* __restrict__ b, float* __restrict__ o,
                     int rows, int D)
{
  long long i = blockIdx.x * 256LL + threadIdx.x;
  if (i >= rows) return;
  const float* r = x + i * (long long)D;
  float m = 0.0f;
  for (int d = 0; d < D; ++d) m += r[d];
  m /= (float)D;
  float v = 0.0f;
  for (int d = 0; d < D; ++d) { float t = r[d] - m; v += t * t; }
  v /= (float)D;
  float inv = 1.0f / sqrtf(v + 1e-5f);
  float* oo = o + i * (long long)D;
  for (int d = 0; d < D; ++d) oo[d] = (r[d] - m) * inv * g[d] + b[d];
}

__global__ void k_gpool(const float* __restrict__ x, float* __restrict__ pT,
                        int C, int HW)
{
  long long i = blockIdx.x * 256LL + threadIdx.x;
  if (i >= 16LL * C) return;
  int c = (int)(i % C); int b = (int)(i / C);
  const float* p = x + ((long long)b * C + c) * HW;
  float s = 0.0f;
  for (int j = 0; j < HW; ++j) s += p[j];
  pT[(long long)c * 16 + b] = s / (float)HW;
}

__global__ void k_outT(const float* __restrict__ C2, float* __restrict__ out, int O)
{
  long long i = blockIdx.x * 256LL + threadIdx.x;
  if (i >= 16LL * O) return;
  int o = (int)(i % O); int b = (int)(i / O);
  out[(long long)b * O + o] = C2[(long long)o * 16 + b];
}

// ===========================================================================
static inline dim3 nb(long long n) { return dim3((unsigned)((n + 255) / 256)); }

static void gemm(const float* A, const float* B, float* C, int M, int N, int K,
                 long long sA, long long sB, long long sC,
                 const float* bias, int bm, const float* g, const float* bt,
                 int act, int batch, hipStream_t s)
{
  dim3 gr((unsigned)((N + GT_N - 1) / GT_N), (unsigned)((M + GT_M - 1) / GT_M),
          (unsigned)batch);
  k_gemm<<<gr, 128, 0, s>>>(A, B, C, M, N, K, sA, sB, sC, bias, bm, g, bt, act);
}

extern "C" void kernel_launch(void* const* d_in, const int* in_sizes, int n_in,
                              void* d_out, int out_size, void* d_ws, size_t ws_size,
                              hipStream_t stream)
{
  (void)in_sizes; (void)n_in; (void)out_size;
  // ------------- workspace plan (float offsets) -------------
  const size_t OFF_X = 0, OFF_X0 = 3211264, OFF_TP = 5619712, ARENA = 5677056;
  const size_t A_BIG = ARENA,              A_H2 = ARENA + 12845056;
  const size_t A_HG  = ARENA + 19726336,   A_XP = ARENA + 26607616;
  const size_t A_ANT = ARENA + 30048256,   A_X1 = ARENA + 33488896;
  const size_t A_Y   = ARENA + 36700160,   A_BN = ARENA + 37732352;
  const size_t A_IDX = ARENA + 38764544,   A_F2 = ARENA + 39248384;
  const size_t NEED  = (ARENA + 42459648) * sizeof(float);
  if (ws_size < NEED) return;   // workspace too small -> no-op (deterministic)

  float* Wf  = (float*)d_ws;
  float* X   = Wf + OFF_X;      // main activation (persistent per stage)
  float* X0  = Wf + OFF_X0;     // prompted image
  float* TP  = Wf + OFF_TP;     // token prompts (B,14,256)
  int*   IDX = (int*)(Wf + A_IDX);

  // ------------- unpack inputs (insertion order, depth-first) -------------
  int p = 0;
  auto nx = [&]() { return (const float*)d_in[p++]; };
  const float* IN = nx();
  const float *m1w = nx(), *m1b = nx(), *m2w = nx(), *m2b = nx();
  const float *n1w = nx(), *n1b = nx(), *n2w = nx(), *n2b = nx();
  const float *c1w = nx(), *c1b = nx(), *c2w = nx(), *c2b = nx();
  const float *c3w = nx(), *c3b = nx(), *hww = nx(), *hwb = nx();
  const float *lng = nx(), *lnb = nx();
  const float *sw1 = nx(), *sb1 = nx(), *sg1 = nx(), *sbt1 = nx();
  const float *sw2 = nx(), *sb2 = nx(), *sg2 = nx(), *sbt2 = nx();
  const float *sw3 = nx(), *sb3 = nx(), *sg3 = nx(), *sbt3 = nx();
  const float *pe  = nx();
  struct StP {
    const float *dw,*db,*dg,*dbt;
    const float *f1w,*f1b,*f1g,*f1bt,*np;
    const float *gcw,*gcb,*gcg,*gcbt;
    const float *f2w,*f2b,*f2g,*f2bt;
    const float *w1,*b1,*g1,*bt1,*w2,*b2,*g2,*bt2;
  } SP[4];
  for (int i = 0; i < 4; ++i) {
    StP& s = SP[i];
    if (i > 0) { s.dw = nx(); s.db = nx(); s.dg = nx(); s.dbt = nx(); }
    else       { s.dw = s.db = s.dg = s.dbt = nullptr; }
    s.f1w = nx(); s.f1b = nx(); s.f1g = nx(); s.f1bt = nx(); s.np = nx();
    s.gcw = nx(); s.gcb = nx(); s.gcg = nx(); s.gcbt = nx();
    s.f2w = nx(); s.f2b = nx(); s.f2g = nx(); s.f2bt = nx();
    s.w1 = nx(); s.b1 = nx(); s.g1 = nx(); s.bt1 = nx();
    s.w2 = nx(); s.b2 = nx(); s.g2 = nx(); s.bt2 = nx();
  }
  const float *hw1 = nx(), *hb1 = nx(), *hg1 = nx(), *hbt1 = nx();
  const float *hw2 = nx(), *hb2 = nx();

  // ================= phase 1: get_prompts =================
  float* PA  = Wf + A_BIG;                 // (3136,768)
  float* PH1 = Wf + A_BIG + 2408448;       // (3136,64)
  float* PP1 = Wf + A_BIG + 2609152;       // (3136,768)
  float* PT1 = Wf + A_H2;                  // (16,16,224,224)
  float* PP2 = Wf + A_XP;                  // (16,3,224,224)
  k_patches<<<nb(3136LL * 768), 256, 0, stream>>>(IN, PA);
  gemm(PA, m1w, PH1, 3136, 64, 768, 0, 0, 0, m1b, 2, nullptr, nullptr, 1, 1, stream);
  gemm(PH1, m2w, PP1, 3136, 768, 64, 0, 0, 0, m2b, 2, nullptr, nullptr, 0, 1, stream);
  k_conv3x3<<<nb(16LL * 16 * 224 * 224), 256, 0, stream>>>(
      IN, n1w, n1b, nullptr, nullptr, PT1, 16, 3, 224, 224, 16, 224, 224, 1, 1);
  k_conv3x3<<<nb(16LL * 3 * 224 * 224), 256, 0, stream>>>(
      PT1, n2w, n2b, nullptr, nullptr, PP2, 16, 16, 224, 224, 3, 224, 224, 1, 0);
  k_make_x0<<<nb(16LL * 3 * 224 * 224), 256, 0, stream>>>(IN, PP1, PP2, X0);

  // ================= phase 2: token_prompt (uses raw image) =================
  float* TA = Wf + A_BIG;                  // (16,14,224,224)
  float* TB = Wf + A_H2;                   // (16,14,52,52)
  float* TC = Wf + A_H2 + 605696;          // (16,42,52,52)
  float* TD = Wf + A_HG;                   // (16,42,16,16)
  float* TE = Wf + A_HG + 172032;          // (16,42,16,16)
  float* TF = Wf + A_XP;                   // (16,14,256)
  k_conv3x3<<<nb(16LL * 14 * 224 * 224), 256, 0, stream>>>(
      IN, c1w, c1b, nullptr, nullptr, TA, 16, 3, 224, 224, 14, 224, 224, 1, 2);
  k_maxpool<<<nb(16LL * 14 * 52 * 52), 256, 0, stream>>>(TA, TB, 16 * 14, 224, 224, 8, 8, 4, 52, 52);
  k_conv3x3<<<nb(16LL * 42 * 52 * 52), 256, 0, stream>>>(
      TB, c2w, c2b, nullptr, nullptr, TC, 16, 14, 52, 52, 42, 52, 52, 1, 2);
  k_maxpool<<<nb(16LL * 42 * 16 * 16), 256, 0, stream>>>(TC, TD, 16 * 42, 52, 52, 2, 2, 3, 16, 16);
  k_conv3x3<<<nb(16LL * 42 * 16 * 16), 256, 0, stream>>>(
      TD, c3w, c3b, nullptr, nullptr, TE, 16, 42, 16, 16, 42, 16, 16, 1, 0);
  gemm(hww, TE, TF, 14, 256, 42, 0, 42LL * 256, 14LL * 256, hwb, 1, nullptr, nullptr, 0, 16, stream);
  k_ln<<<nb(16 * 14), 256, 0, stream>>>(TF, lng, lnb, TP, 16 * 14, 256);

  // ================= phase 3: stem =================
  float* ST1 = Wf + A_BIG;                 // (16,32,112,112)
  float* ST2 = Wf + A_H2;                  // (16,64,56,56)
  k_conv3x3<<<nb(16LL * 32 * 112 * 112), 256, 0, stream>>>(
      X0, sw1, sb1, sg1, sbt1, ST1, 16, 3, 224, 224, 32, 112, 112, 2, 1);
  k_conv3x3<<<nb(16LL * 64 * 56 * 56), 256, 0, stream>>>(
      ST1, sw2, sb2, sg2, sbt2, ST2, 16, 32, 112, 112, 64, 56, 56, 2, 1);
  k_conv3x3<<<nb(16LL * 64 * 56 * 56), 256, 0, stream>>>(
      ST2, sw3, sb3, sg3, sbt3, X, 16, 64, 56, 56, 64, 56, 56, 1, 0);
  k_addpe<<<nb(16LL * 64 * 56 * 56), 256, 0, stream>>>(X, pe, 16LL * 64 * 56 * 56, 64LL * 56 * 56);

  // ================= phase 4: stages =================
  const int Cc[4] = {64, 128, 256, 512};
  const int Hs[4] = {56, 28, 14, 7};
  const int Rr[4] = {4, 2, 1, 1};
  for (int i = 0; i < 4; ++i) {
    const StP& s = SP[i];
    const int C = Cc[i], H = Hs[i], Wd = H, r = Rr[i];
    if (i > 0) {
      const int Cp = Cc[i - 1], Hp_ = Hs[i - 1];
      float* DT = Wf + A_F2;
      k_conv3x3<<<nb(16LL * C * H * Wd), 256, 0, stream>>>(
          X, s.dw, s.db, s.dg, s.dbt, DT, 16, Cp, Hp_, Hp_, C, H, Wd, 2, 0);
      hipMemcpyAsync(X, DT, (size_t)16 * C * H * Wd * sizeof(float),
                     hipMemcpyDeviceToDevice, stream);
    }
    const int HW = H * Wd, Hp = H + 4, Nn = Hp * Wd;
    const int Mm = (r > 1) ? (Hp / r) * (Wd / r) : Nn;
    float* X1  = Wf + A_X1;   // fc1 out (B,C,HW); later reused as xg
    float* XP  = Wf + A_XP;   // (B,C,Nn)
    float* Yb  = Wf + A_Y;    // (B,C,Mm) raw pooled
    float* ANT = Wf + A_ANT;  // (B,Nn,C) normalized; later reused as X2
    float* BN  = Wf + A_BN;   // (B,C,Mm) normalized
    float* Sm  = Wf + A_BIG;  // (B,Nn,Mm); later reused as F1
    float* H2  = Wf + A_H2;   // (B,2C,Nn)
    float* HG  = Wf + A_HG;   // (B,2C,Nn)
    float* F2  = Wf + A_F2;   // (B,C,HW)

    // grapher
    gemm(s.f1w, X, X1, C, HW, C, 0, (long long)C * HW, (long long)C * HW,
         s.f1b, 1, s.f1g, s.f1bt, 0, 16, stream);
    k_concat<<<nb(16LL * C * Nn), 256, 0, stream>>>(X1, s.np, TP, XP, C, H, Wd, C == 256);
    const float* Yraw;
    if (r > 1) {
      k_avgpool<<<nb(16LL * C * Mm), 256, 0, stream>>>(XP, Yb, 16 * C, Hp, Wd, r, Hp / r, Wd / r);
      Yraw = Yb;
    } else {
      Yraw = XP;
    }
    k_l2t<<<nb(16LL * Nn), 256, 0, stream>>>(XP, ANT, C, Nn);
    k_l2 <<<nb(16LL * Mm), 256, 0, stream>>>(Yraw, BN, C, Mm);
    gemm(ANT, BN, Sm, Nn, Mm, C, (long long)Nn * C, (long long)C * Mm,
         (long long)Nn * Mm, nullptr, 0, nullptr, nullptr, 0, 16, stream);
    k_topk<<<nb(16LL * Nn), 256, 0, stream>>>(Sm, IDX, Nn, Mm);
    k_grmax<<<nb(16LL * C * Nn), 256, 0, stream>>>(XP, Yraw, IDX, H2, C, Nn, Mm);
    gemm(s.gcw, H2, HG, 2 * C, Nn, 2 * C, 0, (long long)2 * C * Nn,
         (long long)2 * C * Nn, s.gcb, 1, s.gcg, s.gcbt, 1, 16, stream);
    float* X2 = ANT;  // reuse
    gemm(s.f2w, HG, X2, C, Nn, 2 * C, 0, (long long)2 * C * Nn,
         (long long)C * Nn, s.f2b, 1, s.f2g, s.f2bt, 0, 16, stream);
    float* XG = X1;   // reuse
    k_cropres<<<nb(16LL * C * HW), 256, 0, stream>>>(X2, X, XG, C, H, Wd);

    // ffn
    float* F1 = Sm;   // reuse big slot
    gemm(s.w1, XG, F1, 4 * C, HW, C, 0, (long long)C * HW, (long long)4 * C * HW,
         s.b1, 1, s.g1, s.bt1, 1, 16, stream);
    gemm(s.w2, F1, F2, C, HW, 4 * C, 0, (long long)4 * C * HW, (long long)C * HW,
         s.b2, 1, s.g2, s.bt2, 0, 16, stream);
    k_add<<<nb(16LL * C * HW), 256, 0, stream>>>(F2, XG, X, 16LL * C * HW);
  }

  // ================= phase 5: head =================
  float* PTt = Wf + A_BIG;            // (512,16)
  float* C1  = Wf + A_BIG + 8192;     // (1024,16)
  float* C2  = Wf + A_BIG + 24576;    // (1000,16)
  k_gpool<<<nb(16LL * 512), 256, 0, stream>>>(X, PTt, 512, 49);
  gemm(hw1, PTt, C1, 1024, 16, 512, 0, 0, 0, hb1, 1, hg1, hbt1, 1, 1, stream);
  gemm(hw2, C1, C2, 1000, 16, 1024, 0, 0, 0, hb2, 1, nullptr, nullptr, 0, 1, stream);
  k_outT<<<nb(16LL * 1000), 256, 0, stream>>>(C2, (float*)d_out, 1000);
}
